// SNNAdaptiveThreshold_50379966382533
// MI455X (gfx1250) — compile-verified, matrix operand order checked by reading, not run on hardware
//
#include <hip/hip_runtime.h>
#include <hip/hip_bf16.h>
#include <stdint.h>
#include <stddef.h>

typedef __attribute__((ext_vector_type(16))) _Float16 v16h;
typedef __attribute__((ext_vector_type(8)))  _Float16 v8h;
typedef __attribute__((ext_vector_type(4)))  _Float16 v4h;
typedef __attribute__((ext_vector_type(8)))  float    v8f;
typedef __attribute__((ext_vector_type(4)))  int      i4v;

#define T_STEPS   100
#define BATCH     256
#define IN_DIM    6400
#define NL1       1024
#define NL2       512
#define NL3       6
#define MROWS     (T_STEPS * BATCH)   // 25600 batched rows

#define DECAY     0.1f
#define TH_DECAY  0.05f
#define TH_PLUS   0.05f

// ---- CDNA5 async global->LDS DMA path (ASYNCcnt), with safe fallback ----
#if defined(__gfx1250__) && __has_builtin(__builtin_amdgcn_global_load_async_to_lds_b128)
#define USE_ASYNC_LDS 1
#else
#define USE_ASYNC_LDS 0
#endif

#define AS1 __attribute__((address_space(1)))
#define AS3 __attribute__((address_space(3)))

__device__ __forceinline__ void async_copy16(const void* g, void* l) {
#if USE_ASYNC_LDS
  // Builtin signature: (AS1 int4* gsrc, AS3 int4* ldst, imm offset, imm cpol).
  // global flat addr == as(1) addr; LDS flat addr low 32 bits == LDS offset (ISA 10.2)
  __builtin_amdgcn_global_load_async_to_lds_b128(
      (AS1 i4v*)(uintptr_t)g,
      (AS3 i4v*)(uint32_t)(uintptr_t)l,
      0, 0);
#else
  (void)g; (void)l;
#endif
}

__device__ __forceinline__ void wait_async0() {
#if USE_ASYNC_LDS
#if __has_builtin(__builtin_amdgcn_s_wait_asynccnt)
  __builtin_amdgcn_s_wait_asynccnt(0);
#else
  asm volatile("s_wait_asynccnt 0x0" ::: "memory");
#endif
#endif
}

// ---------------- f32 -> f16 weight conversion ----------------
__global__ void cvt_f32_f16(const float* __restrict__ src, _Float16* __restrict__ dst, int n) {
  int i = blockIdx.x * blockDim.x + threadIdx.x;
  if (i < n) dst[i] = (_Float16)src[i];
}

// ---------------- WMMA GEMM: C[M,N] = A[M,K] * B[N,K]^T ----------------
// Block tile 128x128, K-step 32, 256 threads = 8 waves arranged 2(M) x 4(N).
// Each wave: 4x2 tiles of v_wmma_f32_16x16x32_f16. Double-buffered LDS.
// f16 tiles ride the async global->LDS DMA (no VGPR staging); int32 spike
// tiles are converted to f16 through VGPRs. One barrier per K-step.
template<bool A_INT>
__global__ __launch_bounds__(256) void gemm_wmma(const void* __restrict__ Av,
                                                 const _Float16* __restrict__ Bmat,
                                                 float* __restrict__ Cmat,
                                                 const int Nn, const int K) {
  constexpr int LDA = 40;   // padded LDS row stride (halfs): 80B rows -> conflict-free frag reads
  __shared__ _Float16 As[2][128 * LDA];
  __shared__ _Float16 Bs[2][128 * LDA];

  const int tid  = threadIdx.x;
  const int lane = tid & 31;
  const int wave = tid >> 5;
  const int wm   = wave >> 2;          // 0..1  (M direction)
  const int wn   = wave & 3;           // 0..3  (N direction)
  const int bm   = blockIdx.y * 128;
  const int bn   = blockIdx.x * 128;
  const int ksteps = K >> 5;

  const int*      Ai = (const int*)Av;
  const _Float16* Ah = (const _Float16*)Av;

  v8f acc[4][2];
  #pragma unroll
  for (int mi = 0; mi < 4; ++mi)
    #pragma unroll
    for (int ni = 0; ni < 2; ++ni)
      #pragma unroll
      for (int j = 0; j < 8; ++j) acc[mi][ni][j] = 0.0f;

  int4 aPreI[4];
#if !USE_ASYNC_LDS
  v8h  aPreH[2];
  v8h  bPre[2];
#endif

  // --- register-staged path (always for int32 A; everything in fallback mode)
  auto fetch = [&](int k0) {
    if constexpr (A_INT) {
      #pragma unroll
      for (int p = 0; p < 4; ++p) {
        const int row = p * 32 + (tid >> 3);
        const int col = (tid & 7) * 4;
        aPreI[p] = *(const int4*)(Ai + (size_t)(bm + row) * K + k0 + col);
      }
    }
#if !USE_ASYNC_LDS
    if constexpr (!A_INT) {
      #pragma unroll
      for (int p = 0; p < 2; ++p) {
        const int row = p * 64 + (tid >> 2);
        const int col = (tid & 3) * 8;
        aPreH[p] = *(const v8h*)(Ah + (size_t)(bm + row) * K + k0 + col);
      }
    }
    #pragma unroll
    for (int p = 0; p < 2; ++p) {
      const int row = p * 64 + (tid >> 2);
      const int col = (tid & 3) * 8;
      bPre[p] = *(const v8h*)(Bmat + (size_t)(bn + row) * K + k0 + col);
    }
#endif
  };

  auto commit = [&](int buf) {
    if constexpr (A_INT) {
      #pragma unroll
      for (int p = 0; p < 4; ++p) {
        const int row = p * 32 + (tid >> 3);
        const int col = (tid & 7) * 4;
        v4h h;
        h[0] = (_Float16)(float)aPreI[p].x;
        h[1] = (_Float16)(float)aPreI[p].y;
        h[2] = (_Float16)(float)aPreI[p].z;
        h[3] = (_Float16)(float)aPreI[p].w;
        *(v4h*)&As[buf][row * LDA + col] = h;
      }
    }
#if !USE_ASYNC_LDS
    if constexpr (!A_INT) {
      #pragma unroll
      for (int p = 0; p < 2; ++p) {
        const int row = p * 64 + (tid >> 2);
        const int col = (tid & 3) * 8;
        *(v8h*)&As[buf][row * LDA + col] = aPreH[p];
      }
    }
    #pragma unroll
    for (int p = 0; p < 2; ++p) {
      const int row = p * 64 + (tid >> 2);
      const int col = (tid & 3) * 8;
      *(v8h*)&Bs[buf][row * LDA + col] = bPre[p];
    }
#endif
  };

  // --- async DMA path: f16 tiles go straight global->LDS, tracked by ASYNCcnt
  auto issue_async = [&](int k0, int buf) {
#if USE_ASYNC_LDS
    if constexpr (!A_INT) {
      #pragma unroll
      for (int p = 0; p < 2; ++p) {
        const int row = p * 64 + (tid >> 2);
        const int col = (tid & 3) * 8;
        async_copy16(Ah + (size_t)(bm + row) * K + k0 + col, &As[buf][row * LDA + col]);
      }
    }
    #pragma unroll
    for (int p = 0; p < 2; ++p) {
      const int row = p * 64 + (tid >> 2);
      const int col = (tid & 3) * 8;
      async_copy16(Bmat + (size_t)(bn + row) * K + k0 + col, &Bs[buf][row * LDA + col]);
    }
#else
    (void)k0; (void)buf;
#endif
  };

  fetch(0);
  issue_async(0, 0);
  commit(0);
  wait_async0();
  __syncthreads();

  int cur = 0;
  for (int kk = 0; kk < ksteps; ++kk) {
    const bool more = (kk + 1 < ksteps);
    if (more) {
      fetch((kk + 1) << 5);
      issue_async((kk + 1) << 5, cur ^ 1);   // nxt buffer: prior readers done at last barrier
    }

    // --- fragment loads per ISA 16-bit A/B layout:
    // lanes 0-15: K 0-7 (v0-3) and 16-23 (v4-7); lanes 16-31: +8.
    const _Float16* Ab = As[cur];
    const _Float16* Bb = Bs[cur];
    const int c0   = (lane >> 4) * 8;   // half offset for this lane half
    const int rsel = lane & 15;

    v16h af[4];
    #pragma unroll
    for (int mi = 0; mi < 4; ++mi) {
      const int row = wm * 64 + mi * 16 + rsel;
      v8h x0 = *(const v8h*)&Ab[row * LDA + c0];
      v8h x1 = *(const v8h*)&Ab[row * LDA + c0 + 16];
      #pragma unroll
      for (int i = 0; i < 8; ++i) { af[mi][i] = x0[i]; af[mi][8 + i] = x1[i]; }
    }
    v16h bf[2];
    #pragma unroll
    for (int ni = 0; ni < 2; ++ni) {
      const int col = wn * 32 + ni * 16 + rsel;
      v8h x0 = *(const v8h*)&Bb[col * LDA + c0];
      v8h x1 = *(const v8h*)&Bb[col * LDA + c0 + 16];
      #pragma unroll
      for (int i = 0; i < 8; ++i) { bf[ni][i] = x0[i]; bf[ni][8 + i] = x1[i]; }
    }

    #pragma unroll
    for (int mi = 0; mi < 4; ++mi)
      #pragma unroll
      for (int ni = 0; ni < 2; ++ni)
        acc[mi][ni] = __builtin_amdgcn_wmma_f32_16x16x32_f16(
            false, af[mi], false, bf[ni], (short)0, acc[mi][ni], false, false);

    if (more) commit(cur ^ 1);
    wait_async0();
    __syncthreads();
    cur ^= 1;
  }

  // --- epilogue: D layout = VGPR j holds row (j + 8*(lane>=16)), col = lane&15
  #pragma unroll
  for (int mi = 0; mi < 4; ++mi) {
    const int rbase = bm + wm * 64 + mi * 16 + ((lane >> 4) << 3);
    #pragma unroll
    for (int ni = 0; ni < 2; ++ni) {
      const int col = bn + wn * 32 + ni * 16 + (lane & 15);
      #pragma unroll
      for (int j = 0; j < 8; ++j)
        Cmat[(size_t)(rbase + j) * Nn + col] = acc[mi][ni][j];
    }
  }
}

// ---------------- LIF scan over time (elementwise recurrence) ----------------
template<int NN>
__global__ __launch_bounds__(256) void lif_scan(const float* __restrict__ I,
                                                const float* __restrict__ bias,
                                                _Float16* __restrict__ Sout) {
  const int idx = blockIdx.x * blockDim.x + threadIdx.x;   // over BATCH*NN
  const float b = bias[idx & (NN - 1)];
  float V = 0.0f, Th = 1.0f;
  const int strideT = BATCH * NN;
  #pragma unroll 4
  for (int t = 0; t < T_STEPS; ++t) {
    float cur = I[(size_t)t * strideT + idx] + b;
    V = V * (1.0f - DECAY) + cur;
    float S = (V >= Th) ? 1.0f : 0.0f;
    V = (S > 0.0f) ? 0.0f : V;
    Th = Th * (1.0f - TH_DECAY) + TH_PLUS * S;
    Sout[(size_t)t * strideT + idx] = (_Float16)S;
  }
}

// ---------------- layer 3: I3[m, 0..5] = S2[m,:] . W3[j,:]  (K=512, N=6) ----------------
__global__ __launch_bounds__(256) void layer3_gemv(const _Float16* __restrict__ S2,
                                                   const float* __restrict__ W3,
                                                   float* __restrict__ I3) {
  __shared__ float Ws[NL3 * NL2];          // 12 KB
  for (int i = threadIdx.x; i < NL3 * NL2; i += 256) Ws[i] = W3[i];
  __syncthreads();

  const int lane = threadIdx.x & 31;
  const int wave = threadIdx.x >> 5;
  const int m = blockIdx.x * 8 + wave;     // 25600 rows / 8 waves

  const _Float16* row = S2 + (size_t)m * NL2 + lane * 16;
  v8h s0 = *(const v8h*)row;
  v8h s1 = *(const v8h*)(row + 8);

  float p[NL3] = {0.f, 0.f, 0.f, 0.f, 0.f, 0.f};
  #pragma unroll
  for (int i = 0; i < 16; ++i) {
    const float s = (float)(i < 8 ? s0[i] : s1[i - 8]);
    const int k = lane * 16 + i;
    #pragma unroll
    for (int j = 0; j < NL3; ++j) p[j] += s * Ws[j * NL2 + k];
  }
  #pragma unroll
  for (int off = 16; off > 0; off >>= 1)
    #pragma unroll
    for (int j = 0; j < NL3; ++j) p[j] += __shfl_xor(p[j], off, 32);

  if (lane == 0) {
    #pragma unroll
    for (int j = 0; j < NL3; ++j) I3[(size_t)m * NL3 + j] = p[j];
  }
}

// ---------------- final LIF scan + spike counting ----------------
__global__ void lif_scan_count(const float* __restrict__ I3, const float* __restrict__ b3,
                               float* __restrict__ out) {
  const int idx = blockIdx.x * blockDim.x + threadIdx.x;   // over BATCH*6
  if (idx >= BATCH * NL3) return;
  const float b = b3[idx % NL3];
  float V = 0.0f, Th = 1.0f, cnt = 0.0f;
  for (int t = 0; t < T_STEPS; ++t) {
    float cur = I3[(size_t)t * BATCH * NL3 + idx] + b;
    V = V * (1.0f - DECAY) + cur;
    float S = (V >= Th) ? 1.0f : 0.0f;
    V = (S > 0.0f) ? 0.0f : V;
    Th = Th * (1.0f - TH_DECAY) + TH_PLUS * S;
    cnt += S;
  }
  out[idx] = cnt;
}

extern "C" void kernel_launch(void* const* d_in, const int* in_sizes, int n_in,
                              void* d_out, int out_size, void* d_ws, size_t ws_size,
                              hipStream_t stream) {
  const int*   spikes = (const int*)d_in[0];   // [100,256,6400] 0/1
  const float* W1 = (const float*)d_in[1];     // [1024,6400]
  const float* b1 = (const float*)d_in[2];
  const float* W2 = (const float*)d_in[3];     // [512,1024]
  const float* b2 = (const float*)d_in[4];
  const float* W3 = (const float*)d_in[5];     // [6,512]
  const float* b3 = (const float*)d_in[6];
  float* out = (float*)d_out;                  // [256,6]

  // Workspace layout (256B-aligned offsets). I2 aliases I1 (stream-ordered).
  char* ws = (char*)d_ws;
  _Float16* W1h = (_Float16*)(ws);                        // 13,107,200 B
  _Float16* W2h = (_Float16*)(ws + 13107200);             //  1,048,576 B
  _Float16* S1  = (_Float16*)(ws + 14155776);             // 52,428,800 B
  _Float16* S2  = (_Float16*)(ws + 66584576);             // 26,214,400 B
  float*    I1  = (float*)   (ws + 92798976);             // 104,857,600 B
  float*    I2  = (float*)   (ws + 92798976);             // alias of I1
  float*    I3  = (float*)   (ws + 197656576);            //    614,400 B

  // 0) weights -> f16 (L2-resident thereafter)
  cvt_f32_f16<<<(NL1 * IN_DIM + 255) / 256, 256, 0, stream>>>(W1, W1h, NL1 * IN_DIM);
  cvt_f32_f16<<<(NL2 * NL1 + 255) / 256, 256, 0, stream>>>(W2, W2h, NL2 * NL1);

  // 1) I1 = spikes @ W1^T   (25600 x 1024, K=6400)  -- dominant GEMM
  gemm_wmma<true><<<dim3(NL1 / 128, MROWS / 128), 256, 0, stream>>>(
      (const void*)spikes, W1h, I1, NL1, IN_DIM);

  // 2) layer-1 LIF scan -> S1 (f16 spikes)
  lif_scan<NL1><<<(BATCH * NL1) / 256, 256, 0, stream>>>(I1, b1, S1);

  // 3) I2 = S1 @ W2^T   (25600 x 512, K=1024)
  gemm_wmma<false><<<dim3(NL2 / 128, MROWS / 128), 256, 0, stream>>>(
      (const void*)S1, W2h, I2, NL2, NL1);

  // 4) layer-2 LIF scan -> S2
  lif_scan<NL2><<<(BATCH * NL2) / 256, 256, 0, stream>>>(I2, b2, S2);

  // 5) I3 = S2 @ W3^T   (25600 x 6, K=512)
  layer3_gemv<<<MROWS / 8, 256, 0, stream>>>(S2, W3, I3);

  // 6) layer-3 LIF scan + spike counts -> out
  lif_scan_count<<<(BATCH * NL3 + 255) / 256, 256, 0, stream>>>(I3, b3, out);
}
